// GAT_78546361909763
// MI455X (gfx1250) — matrix operand order repaired and checked by reading, compiled, not run on hardware
//
#include <hip/hip_runtime.h>
#include <hip/hip_bf16.h>

#define BB 16
#define NN 2048
#define DD 256
#define SPLIT 8
#define EPSV 1e-14f
#define SLOPE 0.2f

typedef __attribute__((ext_vector_type(16))) __bf16 v16bf;
typedef __attribute__((ext_vector_type(8)))  float  v8f;
typedef __attribute__((ext_vector_type(8)))  unsigned short ushort8;

union Frag { v16bf v; ushort8 h[2]; };

__device__ inline unsigned short f2bf(float f) {
  unsigned int x = __float_as_uint(f);
  unsigned int r = x + 0x7FFFu + ((x >> 16) & 1u);  // round-to-nearest-even
  return (unsigned short)(r >> 16);
}

// ---------------- Kernel 1: u[b,i] = x.w_hi (s_i), v[b,j] = x.w_lo (s_j) ----
__global__ void __launch_bounds__(256) k_rowdots(
    const float* __restrict__ x, const float* __restrict__ w,
    float* __restrict__ u, float* __restrict__ v) {
  int t = threadIdx.x;
  int lane = t & 31;
  int row = blockIdx.x * 8 + (t >> 5);          // 0 .. B*N-1
  const float* xr = x + (size_t)row * DD;
  float pj = 0.f, pi = 0.f;
  #pragma unroll
  for (int k = lane; k < DD; k += 32) {
    float xv = xr[k];
    pj = fmaf(xv, w[k], pj);        // weight[:D]  -> column term s_j
    pi = fmaf(xv, w[DD + k], pi);   // weight[D:]  -> row term    s_i
  }
  #pragma unroll
  for (int off = 16; off > 0; off >>= 1) {
    pj += __shfl_xor(pj, off, 32);
    pi += __shfl_xor(pi, off, 32);
  }
  if (lane == 0) { v[row] = pj; u[row] = pi; }
}

// ---------------- Kernel 2: per-batch LayerNorm scalars --------------------
__global__ void __launch_bounds__(256) k_stats(
    const float* __restrict__ u, const float* __restrict__ v,
    float* __restrict__ stats) {
  int b = blockIdx.x, t = threadIdx.x;
  float su = 0, suu = 0, sv = 0, svv = 0;
  for (int i = t; i < NN; i += 256) {
    float a = u[b * NN + i], c = v[b * NN + i];
    su += a; suu += a * a; sv += c; svv += c * c;
  }
  #pragma unroll
  for (int off = 16; off > 0; off >>= 1) {
    su  += __shfl_xor(su,  off, 32); suu += __shfl_xor(suu, off, 32);
    sv  += __shfl_xor(sv,  off, 32); svv += __shfl_xor(svv, off, 32);
  }
  __shared__ float sh[8][4];
  int w = t >> 5;
  if ((t & 31) == 0) { sh[w][0] = su; sh[w][1] = suu; sh[w][2] = sv; sh[w][3] = svv; }
  __syncthreads();
  if (t == 0) {
    float a = 0, aa = 0, c = 0, cc = 0;
    for (int i = 0; i < 8; ++i) { a += sh[i][0]; aa += sh[i][1]; c += sh[i][2]; cc += sh[i][3]; }
    float mu = a / NN, mc = c / NN;
    float var = (aa / NN - mu * mu) + (cc / NN - mc * mc); // var over (N,N) grid
    stats[b * 2 + 0] = mu + mc;                 // combined mean
    stats[b * 2 + 1] = rsqrtf(var + EPSV);      // rstd
  }
}

// ---------------- Kernel 3: one-time x -> bf16 transpose (b,d,j)-major -----
__global__ void __launch_bounds__(256) k_xpose(
    const float* __restrict__ x, unsigned short* __restrict__ xT) {
  __shared__ float tile[64][65];
  int b = blockIdx.z;
  int j0 = blockIdx.x * 64, d0 = blockIdx.y * 64;
  int t = threadIdx.x;
  int c = t & 63, rbase = t >> 6;               // 0..3
  #pragma unroll
  for (int r = 0; r < 16; ++r) {
    int jj = rbase + r * 4;                     // coalesced global row reads
    tile[jj][c] = x[((size_t)(b * NN + j0 + jj)) * DD + d0 + c];
  }
  __syncthreads();
  #pragma unroll
  for (int r = 0; r < 16; ++r) {
    int dd = rbase + r * 4;                     // coalesced global writes over j
    xT[((size_t)(b * DD + d0 + dd)) * NN + j0 + c] = f2bf(tile[c][dd]);
  }
}

// ---------------- Kernel 4: partial column softmax stats (split over i) ----
__global__ void __launch_bounds__(256) k_colpart(
    const float* __restrict__ u, const float* __restrict__ v,
    const float* __restrict__ stats,
    const float* __restrict__ gamma, const float* __restrict__ beta,
    float* __restrict__ pm, float* __restrict__ pz) {
  int j = blockIdx.x * 256 + threadIdx.x;
  int b = blockIdx.y;
  int i0 = blockIdx.z * (NN / SPLIT);
  float mt = stats[b * 2], rs = stats[b * 2 + 1];
  float base = (v[b * NN + j] - mt) * rs;       // (u_i+v_j-mt)*rs = u_i*rs + base
  float m = -3.4e38f, Z = 0.f;
  for (int i = i0; i < i0 + NN / SPLIT; ++i) {
    if (i + 16 < NN) {                          // guarded: speculative row prefetch
      __builtin_prefetch(&gamma[(size_t)(i + 16) * NN + j], 0, 1);
      __builtin_prefetch(&beta [(size_t)(i + 16) * NN + j], 0, 1);
    }
    float n = fmaf(u[b * NN + i], rs, base);    // u load is wave-uniform -> s_load
    float val = fmaf(gamma[(size_t)i * NN + j], n, beta[(size_t)i * NN + j]);
    val = (val >= 0.f) ? val : SLOPE * val;     // LeakyReLU
    if (val > m) {                              // rare after warmup: 1 exp common path
      Z = fmaf(Z, __expf(m - val), 1.f);
      m = val;
    } else {
      Z += __expf(val - m);
    }
  }
  size_t idx = ((size_t)blockIdx.z * BB + b) * NN + j;
  pm[idx] = m; pz[idx] = Z;
}

// ---------------- Kernel 4b: combine splits -> m_j, 1/Z_j ------------------
__global__ void __launch_bounds__(256) k_colcombine(
    const float* __restrict__ pm, const float* __restrict__ pz,
    float* __restrict__ colm, float* __restrict__ colinvz) {
  int id = blockIdx.x * 256 + threadIdx.x;      // flat b*N + j
  float m = -3.4e38f;
  #pragma unroll
  for (int s = 0; s < SPLIT; ++s) m = fmaxf(m, pm[(size_t)s * BB * NN + id]);
  float Z = 0.f;
  #pragma unroll
  for (int s = 0; s < SPLIT; ++s)
    Z += pz[(size_t)s * BB * NN + id] * __expf(pm[(size_t)s * BB * NN + id] - m);
  colm[id] = m;
  colinvz[id] = __builtin_amdgcn_rcpf(Z);
}

// ---------------- Kernel 5: flash-style P@x with WMMA bf16 -----------------
// WG tile: 64 rows(i) x 256 cols(d). 8 waves in 2(i) x 4(d) grid, each wave
// owns 32x64 = 2x4 WMMA accumulators. K-loop: j in steps of 32. A (P tile)
// staged in LDS (bf16-pair packed b32 stores); B read straight from global
// bf16 x^T (two contiguous b128 chunks per lane, L2-resident).
__global__ void __launch_bounds__(256) k_attn_mm(
    const unsigned short* __restrict__ xT, const float* __restrict__ u,
    const float* __restrict__ v, const float* __restrict__ stats,
    const float* __restrict__ gamma, const float* __restrict__ beta,
    const float* __restrict__ colm, const float* __restrict__ colinvz,
    float* __restrict__ out) {
  // rows padded to 48 ushorts (96 B): 16-B fragment chunks stay aligned,
  // non-power-of-2 bank stride.
  __shared__ __align__(16) unsigned short Pt[64][48];   // P tile, [i][j] bf16

  int t = threadIdx.x;
  int b = blockIdx.y;
  int i0 = blockIdx.x * 64;
  float mt = stats[b * 2], rs = stats[b * 2 + 1];
  int lane = t & 31, w = t >> 5;
  int wi = w >> 2, wd = w & 3;

  // staging coords: thread t owns column pair jj = (t&15)*2 for rows
  // ii = r*16 + (t>>4), r = 0..3  -> hoist all per-column loads out of r loop
  int jj_s = (t & 15) * 2;
  int ii_s = t >> 4;

  v8f acc[2][4];
  #pragma unroll
  for (int m2 = 0; m2 < 2; ++m2)
    #pragma unroll
    for (int nt = 0; nt < 4; ++nt) acc[m2][nt] = {};

  for (int j0 = 0; j0 < NN; j0 += 32) {
    int j = j0 + jj_s;
    float2 vj = *(const float2*)&v[b * NN + j];
    float2 cm = *(const float2*)&colm[b * NN + j];
    float2 cz = *(const float2*)&colinvz[b * NN + j];
    float b0 = (vj.x - mt) * rs, b1 = (vj.y - mt) * rs;

    __syncthreads();
    // ---- stage P tile: 4 bf16 pairs per thread, packed b32 DS stores
    #pragma unroll
    for (int r = 0; r < 4; ++r) {
      int ii = r * 16 + ii_s;
      int i = i0 + ii;
      float2 g  = *(const float2*)&gamma[(size_t)i * NN + j];
      float2 be = *(const float2*)&beta [(size_t)i * NN + j];
      if (j0 + 64 <= NN)                        // prefetch next K-tile's columns
        __builtin_prefetch(&gamma[(size_t)i * NN + j + 32], 0, 1);
      float uirs = u[b * NN + i] * rs;
      float v0 = fmaf(g.x, uirs + b0, be.x);
      float v1 = fmaf(g.y, uirs + b1, be.y);
      v0 = (v0 >= 0.f) ? v0 : SLOPE * v0;
      v1 = (v1 >= 0.f) ? v1 : SLOPE * v1;
      float p0 = __expf(v0 - cm.x) * cz.x;
      float p1 = __expf(v1 - cm.y) * cz.y;
      unsigned pk = (unsigned)f2bf(p0) | ((unsigned)f2bf(p1) << 16);
      *(unsigned*)&Pt[ii][jj_s] = pk;
    }
    __syncthreads();

    // ---- A fragments from LDS in contiguous k order (K = khalf*16 + s), same as B
    Frag a[2];
    int khalf = lane >> 4;
    #pragma unroll
    for (int m2 = 0; m2 < 2; ++m2) {
      int ii = wi * 32 + m2 * 16 + (lane & 15);
      a[m2].h[0] = *(const ushort8*)&Pt[ii][khalf * 16];      // K = khalf*16 + 0..7
      a[m2].h[1] = *(const ushort8*)&Pt[ii][khalf * 16 + 8];  // K = khalf*16+8+0..7
    }
    // ---- B fragments straight from global x^T: lane = column d, K = j contig
    Frag bfv[4];
    int kbase = khalf * 16;
    #pragma unroll
    for (int nt = 0; nt < 4; ++nt) {
      int dd = wd * 64 + nt * 16 + (lane & 15);
      const ushort8* src =
          (const ushort8*)&xT[((size_t)(b * DD + dd)) * NN + j0 + kbase];
      bfv[nt].h[0] = src[0];
      bfv[nt].h[1] = src[1];
    }
    #pragma unroll
    for (int m2 = 0; m2 < 2; ++m2)
      #pragma unroll
      for (int nt = 0; nt < 4; ++nt)
        acc[m2][nt] = __builtin_amdgcn_wmma_f32_16x16x32_bf16(
            false, a[m2].v, false, bfv[nt].v, (short)0, acc[m2][nt], false, false);
  }

  // ---- epilogue: fast sigmoid (v_rcp_f32), scatter per C/D layout
  #pragma unroll
  for (int m2 = 0; m2 < 2; ++m2)
    #pragma unroll
    for (int nt = 0; nt < 4; ++nt)
      #pragma unroll
      for (int r = 0; r < 8; ++r) {
        int M = (lane >> 4) * 8 + r;
        int Nc = lane & 15;
        int i = i0 + wi * 32 + m2 * 16 + M;
        int d = wd * 64 + nt * 16 + Nc;
        float val = acc[m2][nt][r];
        out[((size_t)(b * NN + i)) * DD + d] =
            __builtin_amdgcn_rcpf(1.f + __expf(-val));
      }
}

// ---------------------------------------------------------------------------
extern "C" void kernel_launch(void* const* d_in, const int* in_sizes, int n_in,
                              void* d_out, int out_size, void* d_ws, size_t ws_size,
                              hipStream_t stream) {
  const float* x     = (const float*)d_in[0];
  const float* wgt   = (const float*)d_in[1];
  const float* gamma = (const float*)d_in[2];
  const float* beta  = (const float*)d_in[3];
  float* out = (float*)d_out;
  float* ws  = (float*)d_ws;

  const int BN = BB * NN;
  const size_t XTF = (size_t)BB * DD * NN / 2;   // bf16 x^T, in float units
  unsigned short* xT = (unsigned short*)ws;      // B*D*N bf16
  float* u     = ws + XTF;                       // B*N
  float* v     = u + BN;                         // B*N
  float* stats = v + BN;                         // 2*B
  float* colm  = stats + 2 * BB;                 // B*N
  float* colz  = colm + BN;                      // B*N (1/Z)
  float* pm    = colz + BN;                      // SPLIT*B*N
  float* pz    = pm + (size_t)SPLIT * BN;        // SPLIT*B*N

  k_rowdots   <<<dim3(BN / 8), 256, 0, stream>>>(x, wgt, u, v);
  k_stats     <<<dim3(BB), 256, 0, stream>>>(u, v, stats);
  k_xpose     <<<dim3(NN / 64, DD / 64, BB), 256, 0, stream>>>(x, xT);
  k_colpart   <<<dim3(NN / 256, BB, SPLIT), 256, 0, stream>>>(u, v, stats, gamma, beta, pm, pz);
  k_colcombine<<<dim3(BN / 256), 256, 0, stream>>>(pm, pz, colm, colz);
  k_attn_mm   <<<dim3(NN / 64, BB), 256, 0, stream>>>(xT, u, v, stats, gamma, beta, colm, colz, out);
}
